// Decoder_46634754900483
// MI455X (gfx1250) — matrix operand order repaired and checked by reading, not compiled, not run on hardware
//
#include <hip/hip_runtime.h>
#include <hip/hip_bf16.h>

#define D_H   64
#define D_G   64
#define ND    20
#define BATCH 16384
#define DEMB  16
#define NFRAG 47

typedef __attribute__((ext_vector_type(16))) _Float16 v16h;
typedef __attribute__((ext_vector_type(8)))  _Float16 v8h;
typedef __attribute__((ext_vector_type(8)))  float    v8f;

#define LDS_FENCE() asm volatile("s_wait_dscnt 0" ::: "memory")

__device__ __forceinline__ float fastrcp(float x) {
#if __has_builtin(__builtin_amdgcn_rcpf)
  return __builtin_amdgcn_rcpf(x);
#else
  return 1.0f / x;
#endif
}
// CDNA5 has V_TANH_F32 in the TRANS unit (co-issues with VALU); fall back to
// the exp-based form if this toolchain doesn't expose a builtin for it.
__device__ __forceinline__ float fast_tanh(float x) {
#if __has_builtin(__builtin_amdgcn_tanhf)
  return __builtin_amdgcn_tanhf(x);
#elif __has_builtin(__builtin_amdgcn_tanh_f32)
  return __builtin_amdgcn_tanh_f32(x);
#else
  float e = __expf(2.0f * x);
  return 1.0f - 2.0f * fastrcp(e + 1.0f);
#endif
}
// sigmoid(x) = 0.5*tanh(x/2) + 0.5  -> 1 TRANS + 1 FMA on gfx1250
__device__ __forceinline__ float fast_sigmoid(float x) {
  return __builtin_fmaf(0.5f, fast_tanh(0.5f * x), 0.5f);
}

// ---- WMMA fragment helpers (A and B both use contiguous k = 16h + s order) ----

// A-matrix f16 16x32: lane<16 -> row=lane, K=k0..k0+15;
// lane>=16 -> row=lane-16, K=k0+16..k0+31.
__device__ __forceinline__ v16h load_a_frag_f32(const float* src, int ld, int k0, int lane) {
  int row = lane & 15;
  int kb  = (lane & 16) ? 16 : 0;
  v16h a;
#pragma unroll
  for (int h = 0; h < 16; ++h) a[h] = (_Float16)src[row * ld + k0 + kb + h];
  return a;
}

// B-matrix f16 32x16: lane<16 -> col=lane, K=k0..k0+15; lane>=16 -> col=lane-16, K=k0+16..k0+31.
__device__ __forceinline__ v16h load_b_frag_f32(const float* w, int ldn, int k0, int n0,
                                                int kmax, int lane) {
  int col = (lane & 15) + n0;
  int kb  = (lane & 16) ? 16 : 0;
  v16h b;
#pragma unroll
  for (int h = 0; h < 16; ++h) {
    int k = k0 + kb + h;
    b[h] = (_Float16)((k < kmax) ? w[k * ldn + col] : 0.0f);
  }
  return b;
}

__device__ __forceinline__ v16h load_b_frag_f16(const _Float16* w, int ldn, int k0, int n0, int lane) {
  int col = (lane & 15) + n0;
  int kb  = (lane & 16) ? 16 : 0;
  v16h b;
#pragma unroll
  for (int h = 0; h < 16; ++h) b[h] = w[(k0 + kb + h) * ldn + col];
  return b;
}

// C/D f32 16x16: lane<16 -> col=lane, rows 0..7 in c[0..7]; lane>=16 -> col=lane-16, rows 8..15.
__device__ __forceinline__ void store_c_frag(float* dst, int ld, int n0, int lane, v8f c) {
  int col = (lane & 15) + n0;
  int rb  = (lane & 16) ? 8 : 0;
#pragma unroll
  for (int r = 0; r < 8; ++r) dst[(rb + r) * ld + col] = c[r];
}

__device__ __forceinline__ v8f wmma16(v16h a, v16h b, v8f c) {
  return __builtin_amdgcn_wmma_f32_16x16x32_f16(false, a, false, b, (short)0, c, false, false);
}

// Fragment index map for the pre-swizzled weight buffer WF[f][lane][16]:
//  f 0..7  : Wc1_h   (n0 = (f>>1)*16, k0 = (f&1)*32)
//  f 8..10 : We      (k0 = (f-8)*32, kmax=66, K padded to 96)
//  f 11..22: Wih     (n0 = (f-11)*16, k0=0, kmax=16, K padded to 32)
//  f 23..46: Whh     (n0 = ((f-23)>>1)*16, k0 = ((f-23)&1)*32)
#define F_WC1H(i, kk)   ((i) * 2 + (kk))
#define F_WE(kk)        (8 + (kk))
#define F_WIH(tile)     (11 + (tile))
#define F_WHH(tile, kk) (23 + (tile) * 2 + (kk))

// ---------------------------------------------------------------------------
// Kernel 0: swizzle all time-invariant weight B-fragments into the exact
// per-lane register image WMMA wants. 48 KB total -> L1/L2 resident.
// ---------------------------------------------------------------------------
__global__ __launch_bounds__(256) void swizzle_weights_kernel(
    const float* __restrict__ Wc1, const float* __restrict__ We,
    const float* __restrict__ Wih, const float* __restrict__ Whh,
    v16h* __restrict__ WF) {
  int wave = threadIdx.x >> 5, lane = threadIdx.x & 31;
  for (int f = wave; f < NFRAG; f += 8) {
    const float* w; int ldn, k0, n0, kmax;
    if (f < 8)       { w = Wc1 + 64 * 64; ldn = 64;  k0 = (f & 1) * 32;        n0 = (f >> 1) * 16;        kmax = 64; }
    else if (f < 11) { w = We;            ldn = 16;  k0 = (f - 8) * 32;        n0 = 0;                    kmax = 66; }
    else if (f < 23) { w = Wih;           ldn = 192; k0 = 0;                   n0 = (f - 11) * 16;        kmax = 16; }
    else             { w = Whh;           ldn = 192; k0 = ((f - 23) & 1) * 32; n0 = ((f - 23) >> 1) * 16; kmax = 64; }
    WF[f * 32 + lane] = load_b_frag_f32(w, ldn, k0, n0, kmax, lane);
  }
}

// ---------------------------------------------------------------------------
// Kernel A: P16 = intermediate @ Wc1[:64] + bc1  (f16), G16 = f16(intermediate)
// Time-invariant half of the compatibility net, hoisted out of the scan.
// ---------------------------------------------------------------------------
__global__ __launch_bounds__(256) void precompute_kernel(
    const float* __restrict__ intermediate, const float* __restrict__ Wc1,
    const float* __restrict__ bc1, _Float16* __restrict__ P16, _Float16* __restrict__ G16) {
  __shared__ _Float16 sW[64 * 64];
  __shared__ float    sb[64];
  int tid = threadIdx.x;
  for (int i = tid; i < 64 * 64; i += 256) sW[i] = (_Float16)Wc1[i];  // rows 0..63 (goal half)
  if (tid < 64) sb[tid] = bc1[tid];
  __syncthreads();

  int wave = tid >> 5, lane = tid & 31;
  int row0 = (blockIdx.x * 8 + wave) * 16;
  const float* src = intermediate + (size_t)row0 * 64;

  // f16 copy of the goal tile (coalesced)
  for (int i = lane; i < 16 * 64; i += 32) G16[(size_t)row0 * 64 + i] = (_Float16)src[i];

  v16h a0 = load_a_frag_f32(src, 64, 0, lane);
  v16h a1 = load_a_frag_f32(src, 64, 32, lane);
  int col = lane & 15, rb = (lane & 16) ? 8 : 0;
  for (int n0 = 0; n0 < 64; n0 += 16) {
    v8f c = {};
    c = wmma16(a0, load_b_frag_f16(sW, 64, 0,  n0, lane), c);
    c = wmma16(a1, load_b_frag_f16(sW, 64, 32, n0, lane), c);
    float bias = sb[n0 + col];
#pragma unroll
    for (int r = 0; r < 8; ++r)
      P16[((size_t)(row0 + rb + r)) * 64 + n0 + col] = (_Float16)(c[r] + bias);
  }
}

// ---------------------------------------------------------------------------
// Kernel B: fused recurrent decoder. 128 threads = 4 independent waves,
// each wave owns a 16-row batch tile and runs all pred_len steps.
// ---------------------------------------------------------------------------
__global__ __launch_bounds__(128) void decode_kernel(
    const int* __restrict__ pred_len_p,
    const float* __restrict__ last_x_rel, const float* __restrict__ zo,
    const _Float16* __restrict__ P16, const _Float16* __restrict__ G16,
    const v16h* __restrict__ WF,
    const float* __restrict__ be,
    const float* __restrict__ bih, const float* __restrict__ bhh,
    const float* __restrict__ Wc2, const float* __restrict__ bc2,
    const float* __restrict__ Wo,  const float* __restrict__ bo,
    float* __restrict__ out) {
  __shared__ float sWc2[64];
  __shared__ float sbih[192], sbhh[192];
  __shared__ float sWo[128], sbo[2], sbe[16], sbc2[1];
  __shared__ float sH[4][16 * 64];    // hidden state, row-major [b][k]
  __shared__ float sWYQ[4][16 * 96];  // cols 0..63: Q then weighted; 64..65: y; 66..95: pad
  __shared__ float sU[4][16 * 32];    // alias: compab/alpha [b*20+n] then emb [b*32+c]

  int tid = threadIdx.x;
  if (tid < 64) sWc2[tid] = Wc2[tid];
  for (int i = tid; i < 192; i += 128) { sbih[i] = bih[i]; sbhh[i] = bhh[i]; }
  if (tid < 128) sWo[tid] = Wo[tid];
  if (tid < 2)   sbo[tid] = bo[tid];
  if (tid < 16)  sbe[tid] = be[tid];
  if (tid == 0)  sbc2[0] = bc2[0];

  int wave = tid >> 5, lane = tid & 31;
  int bb = blockIdx.x * 64 + wave * 16;
  float* H  = sH[wave];
  float* WY = sWYQ[wave];
  float* AL = sU[wave];
  float* EM = sU[wave];

  int col = lane & 15, rb8 = (lane & 16) ? 8 : 0;

  for (int i = lane; i < 16 * 64; i += 32) H[i] = zo[(size_t)bb * 64 + i];
  for (int i = lane; i < 16 * 96; i += 32) {
    int r = i / 96, c = i % 96;
    WY[i] = (c == 64) ? last_x_rel[(bb + r) * 2 + 0]
          : (c == 65) ? last_x_rel[(bb + r) * 2 + 1] : 0.0f;
  }
  // Register-resident C-layout copy of h (persists across timesteps).
  v8f hc[4];
#pragma unroll
  for (int j = 0; j < 4; ++j)
#pragma unroll
    for (int r = 0; r < 8; ++r)
      hc[j][r] = zo[(size_t)(bb + rb8 + r) * 64 + j * 16 + col];
  __syncthreads();

  int PL = pred_len_p[0];

  for (int t = 0; t < PL; ++t) {
    // ---- Q = h @ Wc1_h (16x64 x 64x64), into WY cols 0..63 ----
    v16h aH0 = load_a_frag_f32(H, 64, 0, lane);
    v16h aH1 = load_a_frag_f32(H, 64, 32, lane);
#pragma unroll
    for (int i = 0; i < 4; ++i) {
      v8f c = {};
      c = wmma16(aH0, WF[F_WC1H(i, 0) * 32 + lane], c);
      c = wmma16(aH1, WF[F_WC1H(i, 1) * 32 + lane], c);
      store_c_frag(WY, 96, i * 16, lane, c);
    }
    LDS_FENCE();

    // ---- compab[n,b] = Wc2 . tanh(P16[n,b,:] + Q[b,:]) + bc2 ----
    for (int i = 0; i < 10; ++i) {
      int tt = i * 32 + lane;
      int n = tt >> 4, b = tt & 15;
      const v8h* p8 = (const v8h*)(P16 + ((size_t)n * BATCH + bb + b) * 64);
      const float* qrow = WY + b * 96;
      float acc = 0.0f;
#pragma unroll
      for (int kk = 0; kk < 8; ++kk) {
        v8h pv = p8[kk];
#pragma unroll
        for (int j = 0; j < 8; ++j)
          acc += fast_tanh((float)pv[j] + qrow[kk * 8 + j]) * sWc2[kk * 8 + j];
      }
      AL[b * 20 + n] = acc + sbc2[0];
    }
    LDS_FENCE();

    // ---- softmax over destinations (lanes 0..15, one batch row each) ----
    if (lane < 16) {
      int b = lane;
      float m = -1e30f;
      for (int n = 0; n < ND; ++n) m = fmaxf(m, AL[b * 20 + n]);
      float e[ND], s = 0.0f;
      for (int n = 0; n < ND; ++n) { e[n] = __expf(AL[b * 20 + n] - m); s += e[n]; }
      float inv = fastrcp(s);
      for (int n = 0; n < ND; ++n) AL[b * 20 + n] = e[n] * inv;
    }
    LDS_FENCE();

    // ---- weighted[b,:] = sum_n alpha[b,n] * goal[n,b,:] (overwrites Q) ----
    {
      int b = lane & 15, half = lane >> 4;
      float acc[32];
#pragma unroll
      for (int kk = 0; kk < 32; ++kk) acc[kk] = 0.0f;
      for (int n = 0; n < ND; ++n) {
        float a = AL[b * 20 + n];
        const v8h* g8 = (const v8h*)(G16 + ((size_t)n * BATCH + bb + b) * 64 + half * 32);
#pragma unroll
        for (int q = 0; q < 4; ++q) {
          v8h gv = g8[q];
#pragma unroll
          for (int j = 0; j < 8; ++j) acc[q * 8 + j] += a * (float)gv[j];
        }
      }
#pragma unroll
      for (int kk = 0; kk < 32; ++kk) WY[b * 96 + half * 32 + kk] = acc[kk];
    }
    LDS_FENCE();

    // ---- emb = [weighted ; y] @ We + be (K=66 padded to 96) ----
    {
      v8f c = {};
      c = wmma16(load_a_frag_f32(WY, 96, 0,  lane), WF[F_WE(0) * 32 + lane], c);
      c = wmma16(load_a_frag_f32(WY, 96, 32, lane), WF[F_WE(1) * 32 + lane], c);
      c = wmma16(load_a_frag_f32(WY, 96, 64, lane), WF[F_WE(2) * 32 + lane], c);
      float bias = sbe[col];
#pragma unroll
      for (int r = 0; r < 8; ++r) EM[(rb8 + r) * 32 + col] = c[r] + bias;
    }
    LDS_FENCE();

    // ---- GRU gates: gi = emb @ Wih (K=16 pad 32), gh = h @ Whh (K=64) ----
    {
      v16h aE = load_a_frag_f32(EM, 32, 0, lane);
#pragma unroll
      for (int j = 0; j < 4; ++j) {
        int cr = j * 16, cz = 64 + j * 16, cn = 128 + j * 16;
        v8f gir = {}, giz = {}, gin = {};
        gir = wmma16(aE, WF[F_WIH(j)     * 32 + lane], gir);
        giz = wmma16(aE, WF[F_WIH(4 + j) * 32 + lane], giz);
        gin = wmma16(aE, WF[F_WIH(8 + j) * 32 + lane], gin);
        v8f ghr = {}, ghz = {}, ghn = {};
        ghr = wmma16(aH0, WF[F_WHH(j, 0)     * 32 + lane], ghr);
        ghr = wmma16(aH1, WF[F_WHH(j, 1)     * 32 + lane], ghr);
        ghz = wmma16(aH0, WF[F_WHH(4 + j, 0) * 32 + lane], ghz);
        ghz = wmma16(aH1, WF[F_WHH(4 + j, 1) * 32 + lane], ghz);
        ghn = wmma16(aH0, WF[F_WHH(8 + j, 0) * 32 + lane], ghn);
        ghn = wmma16(aH1, WF[F_WHH(8 + j, 1) * 32 + lane], ghn);
#pragma unroll
        for (int r = 0; r < 8; ++r) {
          float rr = fast_sigmoid(gir[r] + sbih[cr + col] + ghr[r] + sbhh[cr + col]);
          float zz = fast_sigmoid(giz[r] + sbih[cz + col] + ghz[r] + sbhh[cz + col]);
          float nn = fast_tanh(gin[r] + sbih[cn + col] + rr * (ghn[r] + sbhh[cn + col]));
          float hnew = (1.0f - zz) * nn + zz * hc[j][r];  // old h from registers
          hc[j][r] = hnew;
          H[(rb8 + r) * 64 + j * 16 + col] = hnew;
        }
      }
    }
    LDS_FENCE();

    // ---- y = h_new @ Wo + bo; emit output, feed back into WY ----
    {
      int b = lane & 15, c2 = lane >> 4;
      float acc = sbo[c2];
      for (int k = 0; k < 64; ++k) acc += H[b * 64 + k] * sWo[k * 2 + c2];
      out[((size_t)t * BATCH + bb + b) * 2 + c2] = acc;
      WY[b * 96 + 64 + c2] = acc;
    }
    LDS_FENCE();
  }
}

extern "C" void kernel_launch(void* const* d_in, const int* in_sizes, int n_in,
                              void* d_out, int out_size, void* d_ws, size_t ws_size,
                              hipStream_t stream) {
  (void)in_sizes; (void)n_in; (void)out_size; (void)ws_size;
  const int*   pred_len = (const int*)  d_in[0];
  const float* last_x   = (const float*)d_in[1];
  const float* zo       = (const float*)d_in[2];
  /* d_in[3] = zg: values unused by the math */
  const float* inter    = (const float*)d_in[4];
  const float* We  = (const float*)d_in[5];
  const float* be  = (const float*)d_in[6];
  const float* Wih = (const float*)d_in[7];
  const float* Whh = (const float*)d_in[8];
  const float* bih = (const float*)d_in[9];
  const float* bhh = (const float*)d_in[10];
  const float* Wc1 = (const float*)d_in[11];
  const float* bc1 = (const float*)d_in[12];
  const float* Wc2 = (const float*)d_in[13];
  const float* bc2 = (const float*)d_in[14];
  const float* Wo  = (const float*)d_in[15];
  const float* bo  = (const float*)d_in[16];
  float* out = (float*)d_out;

  _Float16* P16 = (_Float16*)d_ws;                        // 20*16384*64 f16 = 41.9 MB
  _Float16* G16 = P16 + (size_t)ND * BATCH * 64;          // +41.9 MB (both L2-resident)
  v16h*     WF  = (v16h*)(G16 + (size_t)ND * BATCH * 64); // +48 KB swizzled weight frags

  swizzle_weights_kernel<<<dim3(1), dim3(256), 0, stream>>>(Wc1, We, Wih, Whh, WF);
  precompute_kernel<<<dim3(ND * BATCH / 128), dim3(256), 0, stream>>>(inter, Wc1, bc1, P16, G16);
  decode_kernel<<<dim3(BATCH / 64), dim3(128), 0, stream>>>(
      pred_len, last_x, zo, P16, G16, WF, be, bih, bhh, Wc2, bc2, Wo, bo, out);
}
